// TransformerBlock_73890617360591
// MI455X (gfx1250) — compile-verified, matrix operand order checked by reading, not run on hardware
//
#include <hip/hip_runtime.h>
#include <math.h>
#include <stdint.h>

// Transformer block fwd for MI455X (gfx1250): bf16 WMMA + TDM-staged GEMMs.
// B=2, T=2048, D=1024, H=16, dk=64, FF=4096.  Requires ~80MB workspace.

#define D_MODEL 1024
#define N_HEADS 16
#define D_HEAD  64
#define D_FF    4096
#define BATCH   2
#define SEQ     2048
#define M_ROWS  (BATCH * SEQ)   // 4096 token rows

typedef __attribute__((ext_vector_type(8)))  __bf16 bf16x8;
typedef __attribute__((ext_vector_type(16))) __bf16 bf16x16;
typedef __attribute__((ext_vector_type(8)))  float  f32x8;
typedef unsigned int u32x4 __attribute__((ext_vector_type(4)));
typedef int          i32x4 __attribute__((ext_vector_type(4)));
typedef int          i32x8 __attribute__((ext_vector_type(8)));

#ifndef __has_builtin
#define __has_builtin(x) 0
#endif
#if __has_builtin(__builtin_amdgcn_tensor_load_to_lds)
#define HAVE_TDM 1
#else
#define HAVE_TDM 0
#endif

static __device__ __forceinline__ unsigned short f32_to_bf16_bits(float f) {
  unsigned u = __float_as_uint(f);
  unsigned r = u + 0x7FFFu + ((u >> 16) & 1u);   // round-to-nearest-even
  return (unsigned short)(r >> 16);
}

// Build a 16-bit WMMA A/B operand (16x32 / 32x16) for one lane.
// ISA layout: lanes 0-15 hold K = [0..7] and [16..23]; lanes 16-31 hold
// K = [8..15] and [24..31].  p points at (row, k0) with contiguous K.
static __device__ __forceinline__ bf16x16 make_frag(const __bf16* p, int half) {
  bf16x8 lo = *(const bf16x8*)(p + half * 8);
  bf16x8 hi = *(const bf16x8*)(p + 16 + half * 8);
  return __builtin_shufflevector(lo, hi, 0,1,2,3,4,5,6,7,8,9,10,11,12,13,14,15);
}

#if HAVE_TDM
// Issue one TDM 2D tile load: tile = 128 rows x 64 cols of bf16 from a
// row-major [tensor_rows, tensor_cols] matrix into LDS at byte offset lds_off.
// D# per CDNA5 ISA ch.8: group0 = {count/flags, lds_addr, global_addr, type=2},
// group1 = {data_size=2B, tensor_dim0/1, tile_dim0/1, tensor_dim0_stride}.
static __device__ __forceinline__ void tdm_load_2d(
    unsigned lds_off, const unsigned short* base, int row0, int col0,
    int tensor_cols, int tensor_rows) {
  const unsigned long long ga =
      (unsigned long long)(size_t)(base + (size_t)row0 * tensor_cols + col0);
  u32x4 g0;
  g0[0] = 1u;                                            // count=1 (valid), user mode
  g0[1] = lds_off;                                       // lds_addr (bytes)
  g0[2] = (unsigned)(ga & 0xFFFFFFFFu);                  // global_addr[31:0]
  g0[3] = (unsigned)((ga >> 32) & 0x01FFFFFFu)           // global_addr[56:32]
        | (2u << 30);                                    // type=2 ("image")
  i32x8 g1;
  g1[0] = (int)(1u << 16);                               // data_size=1 -> 2 bytes
  g1[1] = (int)(((unsigned)tensor_cols & 0xFFFFu) << 16);        // tensor_dim0 lo16
  g1[2] = (int)((((unsigned)tensor_cols >> 16) & 0xFFFFu)        // tensor_dim0 hi16
        | (((unsigned)tensor_rows & 0xFFFFu) << 16));            // tensor_dim1 lo16
  g1[3] = (int)((((unsigned)tensor_rows >> 16) & 0xFFFFu)        // tensor_dim1 hi16
        | (64u << 16));                                          // tile_dim0 = 64
  g1[4] = 128;                                           // tile_dim1=128, tile_dim2=0
  g1[5] = tensor_cols;                                   // tensor_dim0_stride[31:0]
  g1[6] = 0;                                             // stride hi + dim1_stride lo
  g1[7] = 0;
  i32x4 z4 = {0, 0, 0, 0};
#if defined(__clang_major__) && __clang_major__ >= 23
  i32x8 z8 = {0, 0, 0, 0, 0, 0, 0, 0};
  __builtin_amdgcn_tensor_load_to_lds(g0, g1, z4, z4, z8, 0);
#else
  __builtin_amdgcn_tensor_load_to_lds(g0, g1, z4, z4, 0);
#endif
}
#else
// Fallback: cooperative 128x64 bf16 tile copy, 256 threads, b128 transfers.
static __device__ __forceinline__ void stage_tile(
    char* dst, const unsigned short* base, int row0, int col0, int ldsrc) {
  const int tid = threadIdx.x;
#pragma unroll
  for (int i = 0; i < 4; ++i) {
    const int idx = i * 256 + tid;       // 1024 8-elem chunks
    const int r  = idx >> 3;             // 0..127
    const int c8 = (idx & 7) * 8;        // 0..56
    *(bf16x8*)(dst + ((size_t)r * 64 + c8) * 2) =
        *(const bf16x8*)(base + (size_t)(row0 + r) * ldsrc + col0 + c8);
  }
}
#endif

// ---------------------------------------------------------------- conversion
__global__ __launch_bounds__(256) void cvt_f32_to_bf16_kernel(
    const float* __restrict__ s, unsigned short* __restrict__ d, int n4) {
  int i = blockIdx.x * 256 + threadIdx.x;
  if (i < n4) {
    float4 v = ((const float4*)s)[i];
    d[4 * i + 0] = f32_to_bf16_bits(v.x);
    d[4 * i + 1] = f32_to_bf16_bits(v.y);
    d[4 * i + 2] = f32_to_bf16_bits(v.z);
    d[4 * i + 3] = f32_to_bf16_bits(v.w);
  }
}

// ---------------------------------------------------------------- rmsnorm
__global__ __launch_bounds__(256) void rmsnorm_bf16_kernel(
    const float* __restrict__ x, const float* __restrict__ gw,
    unsigned short* __restrict__ out) {
  const int row = blockIdx.x;
  const int tid = threadIdx.x;
  const float4 xv = *(const float4*)(x + (size_t)row * D_MODEL + tid * 4);
  float ss = xv.x * xv.x + xv.y * xv.y + xv.z * xv.z + xv.w * xv.w;
#pragma unroll
  for (int m = 16; m >= 1; m >>= 1) ss += __shfl_xor(ss, m, 32);
  __shared__ float red[8];
  if ((tid & 31) == 0) red[tid >> 5] = ss;
  __syncthreads();
  float tot = 0.f;
#pragma unroll
  for (int i = 0; i < 8; ++i) tot += red[i];
  const float scale = rsqrtf(tot * (1.0f / D_MODEL) + 1e-5f);
  const float4 wv = *(const float4*)(gw + tid * 4);
  unsigned short* o = out + (size_t)row * D_MODEL + tid * 4;
  o[0] = f32_to_bf16_bits(xv.x * wv.x * scale);
  o[1] = f32_to_bf16_bits(xv.y * wv.y * scale);
  o[2] = f32_to_bf16_bits(xv.z * wv.z * scale);
  o[3] = f32_to_bf16_bits(xv.w * wv.w * scale);
}

// ---------------------------------------------------------------- GEMM core
// Per-wave 64x32 MACC from LDS tiles (row-major 128x64 bf16, row stride 64).
static __device__ __forceinline__ void mma_tile(
    const char* abuf, const char* bbuf, int am, int an, int lrow, int half,
    f32x8 (&acc)[4][2]) {
  const __bf16* Al = (const __bf16*)abuf;
  const __bf16* Bl = (const __bf16*)bbuf;
  bf16x16 af[2][4], bfr[2][2];
#pragma unroll
  for (int kd = 0; kd < 2; ++kd) {
#pragma unroll
    for (int i = 0; i < 4; ++i)
      af[kd][i] = make_frag(Al + (am + i * 16 + lrow) * 64 + kd * 32, half);
#pragma unroll
    for (int j = 0; j < 2; ++j)
      bfr[kd][j] = make_frag(Bl + (an + j * 16 + lrow) * 64 + kd * 32, half);
  }
#pragma unroll
  for (int kd = 0; kd < 2; ++kd)
#pragma unroll
    for (int i = 0; i < 4; ++i)
#pragma unroll
      for (int j = 0; j < 2; ++j)
        acc[i][j] = __builtin_amdgcn_wmma_f32_16x16x32_bf16(
            false, af[kd][i], false, bfr[kd][j], (short)0, acc[i][j], false, false);
}

// C[M,N] = A[M,K] (bf16 row-major) * W[N,K]^T (bf16 row-major), i.e. x @ W.T
// Block = 256 threads (8 waves) -> 128x128 C tile; K staged in 64-wide LDS
// tiles by the Tensor Data Mover (double buffered, TENSORcnt-synchronized).
// EPI: 0 = store bf16, 1 = exact-erf GELU then bf16, 2 = fp32 + residual add
template <int EPI>
__global__ __launch_bounds__(256) void gemm_bf16_kernel(
    const unsigned short* __restrict__ Abits,
    const unsigned short* __restrict__ Wbits, int M, int N, int K,
    unsigned short* __restrict__ outB, float* __restrict__ outF,
    const float* __restrict__ resid) {
  extern __shared__ char smem[];   // 64KB: A tiles @0,16K ; B tiles @32K,48K
  const int lane = threadIdx.x & 31;
  const int w    = threadIdx.x >> 5;
  const int half = lane >> 4;
  const int lrow = lane & 15;
  const int bm = blockIdx.x * 128, bn = blockIdx.y * 128;
  const int am = (w & 1) * 64;
  const int an = (w >> 1) * 32;

  f32x8 acc[4][2] = {};

#if HAVE_TDM
  if (w == 0) {   // wave 0 drives the TDM; EXEC is ignored by tensor ops
    tdm_load_2d(0u, Abits, bm, 0, K, M);
    tdm_load_2d(32768u, Wbits, bn, 0, K, N);
  }
  int buf = 0;
  for (int kk = 0; kk < K; kk += 64) {
    if (w == 0) {
      if (kk + 64 < K) {  // prefetch next K-slab into the other buffer
        tdm_load_2d((unsigned)((buf ^ 1) * 16384), Abits, bm, kk + 64, K, M);
        tdm_load_2d((unsigned)(32768 + (buf ^ 1) * 16384), Wbits, bn, kk + 64, K, N);
        __builtin_amdgcn_s_wait_tensorcnt(2);  // current slab landed (in-order)
      } else {
        __builtin_amdgcn_s_wait_tensorcnt(0);
      }
    }
    __syncthreads();
    mma_tile(smem + buf * 16384, smem + 32768 + buf * 16384, am, an, lrow, half, acc);
    __syncthreads();   // readers done before this buffer becomes a TDM target
    buf ^= 1;
  }
#else
  for (int kk = 0; kk < K; kk += 64) {
    __syncthreads();
    stage_tile(smem, Abits, bm, kk, K);
    stage_tile(smem + 32768, Wbits, bn, kk, K);
    __syncthreads();
    mma_tile(smem, smem + 32768, am, an, lrow, half, acc);
  }
#endif

#pragma unroll
  for (int i = 0; i < 4; ++i)
#pragma unroll
    for (int j = 0; j < 2; ++j)
#pragma unroll
      for (int e = 0; e < 8; ++e) {
        const int row = bm + am + i * 16 + half * 8 + e;   // C layout: VGPR e, half
        const int col = bn + an + j * 16 + lrow;
        const float v = acc[i][j][e];
        if (EPI == 0) {
          outB[(size_t)row * N + col] = f32_to_bf16_bits(v);
        } else if (EPI == 1) {
          const float g = 0.5f * v * (1.0f + erff(v * 0.70710678118654752f));
          outB[(size_t)row * N + col] = f32_to_bf16_bits(g);
        } else {
          const size_t o = (size_t)row * N + col;
          outF[o] = v + resid[o];
        }
      }
}

// ---------------------------------------------------------------- attention
// Causal flash attention.  Block = (b,h) x 128 query rows, 8 waves x 16 rows.
// Q/K/V/O all bf16 [B*T, D] row-major, head slice = cols [h*64, h*64+64).
__global__ __launch_bounds__(256) void attn_kernel(
    const unsigned short* __restrict__ Qb, const unsigned short* __restrict__ Kb,
    const unsigned short* __restrict__ Vb, unsigned short* __restrict__ Ob) {
  __shared__ __align__(16) unsigned short Kt[32][72];      // K tile  (keys x d)
  __shared__ __align__(16) unsigned short Vt[64][40];      // V^T tile (d x keys)
  __shared__ __align__(16) unsigned short Pw[8][16][40];   // per-wave P (q x keys)

  const int tid  = threadIdx.x;
  const int lane = tid & 31, w = tid >> 5;
  const int half = lane >> 4, lrow = lane & 15;
  const int b = blockIdx.y / N_HEADS, h = blockIdx.y % N_HEADS;
  const int qbase = blockIdx.x * 128;
  const int q0 = qbase + w * 16;

  const __bf16* Qh = (const __bf16*)Qb + (size_t)b * SEQ * D_MODEL + h * D_HEAD;
  const __bf16* Kh = (const __bf16*)Kb + (size_t)b * SEQ * D_MODEL + h * D_HEAD;
  const unsigned short* Vh = Vb + (size_t)b * SEQ * D_MODEL + h * D_HEAD;

  bf16x16 qf[2];
#pragma unroll
  for (int kd = 0; kd < 2; ++kd)
    qf[kd] = make_frag(Qh + (size_t)(q0 + lrow) * D_MODEL + kd * 32, half);

  float m_i[8], l_i[8];
#pragma unroll
  for (int e = 0; e < 8; ++e) { m_i[e] = -3.0e38f; l_i[e] = 0.f; }
  f32x8 oacc[4] = {};

  const int nkb = qbase / 32 + 4;        // causal coverage for this block
  const int r   = tid >> 3;              // 0..31 key row
  const int c8  = (tid & 7) * 8;         // 8-elem column chunk

  for (int kbi = 0; kbi < nkb; ++kbi) {
    const int kb = kbi * 32;
    __syncthreads();  // previous iteration's readers done
    *(bf16x8*)&Kt[r][c8] = *(const bf16x8*)(Kh + (size_t)(kb + r) * D_MODEL + c8);
    uint4 vv = *(const uint4*)(Vh + (size_t)(kb + r) * D_MODEL + c8);
    unsigned vals[4] = {vv.x, vv.y, vv.z, vv.w};
#pragma unroll
    for (int e = 0; e < 4; ++e) {
      Vt[c8 + 2 * e][r]     = (unsigned short)(vals[e] & 0xFFFFu);
      Vt[c8 + 2 * e + 1][r] = (unsigned short)(vals[e] >> 16);
    }
    __syncthreads();

    if (kb <= q0 + 15) {      // wave-uniform causal skip
      f32x8 s[2] = {};
#pragma unroll
      for (int kd = 0; kd < 2; ++kd)
#pragma unroll
        for (int j = 0; j < 2; ++j) {
          bf16x16 kf = make_frag((const __bf16*)&Kt[j * 16 + lrow][kd * 32], half);
          s[j] = __builtin_amdgcn_wmma_f32_16x16x32_bf16(
              false, qf[kd], false, kf, (short)0, s[j], false, false);
        }
#pragma unroll
      for (int j = 0; j < 2; ++j) {
        const int key = kb + j * 16 + lrow;
#pragma unroll
        for (int e = 0; e < 8; ++e) {
          const float v = s[j][e] * 0.125f;          // 1/sqrt(64)
          const int qrow = q0 + half * 8 + e;
          s[j][e] = (key <= qrow) ? v : -3.0e38f;
        }
      }
#pragma unroll
      for (int e = 0; e < 8; ++e) {
        float rmax = fmaxf(s[0][e], s[1][e]);
#pragma unroll
        for (int m = 8; m >= 1; m >>= 1) rmax = fmaxf(rmax, __shfl_xor(rmax, m, 32));
        const float mnew = fmaxf(m_i[e], rmax);
        const float p0 = __expf(s[0][e] - mnew);
        const float p1 = __expf(s[1][e] - mnew);
        float psum = p0 + p1;
#pragma unroll
        for (int m = 8; m >= 1; m >>= 1) psum += __shfl_xor(psum, m, 32);
        const float corr = __expf(m_i[e] - mnew);
        l_i[e] = l_i[e] * corr + psum;
        m_i[e] = mnew;
#pragma unroll
        for (int n = 0; n < 4; ++n) oacc[n][e] *= corr;
        Pw[w][half * 8 + e][lrow]      = f32_to_bf16_bits(p0);
        Pw[w][half * 8 + e][16 + lrow] = f32_to_bf16_bits(p1);
      }
      bf16x16 pf = make_frag((const __bf16*)&Pw[w][lrow][0], half);
#pragma unroll
      for (int n = 0; n < 4; ++n) {
        bf16x16 vf = make_frag((const __bf16*)&Vt[n * 16 + lrow][0], half);
        oacc[n] = __builtin_amdgcn_wmma_f32_16x16x32_bf16(
            false, pf, false, vf, (short)0, oacc[n], false, false);
      }
    }
  }
  unsigned short* Oh = Ob + (size_t)b * SEQ * D_MODEL + h * D_HEAD;
#pragma unroll
  for (int n = 0; n < 4; ++n)
#pragma unroll
    for (int e = 0; e < 8; ++e) {
      const int trow = q0 + half * 8 + e;
      Oh[(size_t)trow * D_MODEL + n * 16 + lrow] =
          f32_to_bf16_bits(oacc[n][e] / l_i[e]);
    }
}

// ---------------------------------------------------------------- launch
extern "C" void kernel_launch(void* const* d_in, const int* in_sizes, int n_in,
                              void* d_out, int out_size, void* d_ws, size_t ws_size,
                              hipStream_t stream) {
  (void)in_sizes; (void)n_in; (void)out_size; (void)ws_size;
  const float* x    = (const float*)d_in[0];
  const float* wq   = (const float*)d_in[1];
  const float* wk   = (const float*)d_in[2];
  const float* wv   = (const float*)d_in[3];
  const float* wo   = (const float*)d_in[4];
  const float* ln1w = (const float*)d_in[5];
  const float* ln2w = (const float*)d_in[6];
  const float* w1   = (const float*)d_in[7];
  const float* w2   = (const float*)d_in[8];
  float* out = (float*)d_out;

  char* ws = (char*)d_ws;
  const size_t MB = 1ull << 20;
  unsigned short* wq_b = (unsigned short*)(ws + 0 * MB);   // 2MB each
  unsigned short* wk_b = (unsigned short*)(ws + 2 * MB);
  unsigned short* wv_b = (unsigned short*)(ws + 4 * MB);
  unsigned short* wo_b = (unsigned short*)(ws + 6 * MB);
  unsigned short* w1_b = (unsigned short*)(ws + 8 * MB);   // 8MB
  unsigned short* w2_b = (unsigned short*)(ws + 16 * MB);  // 8MB
  unsigned short* Abuf = (unsigned short*)(ws + 24 * MB);  // bf16 4096x1024
  float*          Ybuf = (float*)(ws + 32 * MB);           // fp32 4096x1024
  unsigned short* Qbuf = (unsigned short*)(ws + 48 * MB);
  unsigned short* Kbuf = (unsigned short*)(ws + 56 * MB);
  unsigned short* Vbuf = (unsigned short*)(ws + 64 * MB);
  unsigned short* Hbuf = (unsigned short*)(ws + 48 * MB);  // 32MB, reuses QKV

  auto cvt = [&](const float* s, unsigned short* d, int n) {
    const int n4 = n / 4;
    cvt_f32_to_bf16_kernel<<<dim3((n4 + 255) / 256), dim3(256), 0, stream>>>(s, d, n4);
  };
  cvt(wq, wq_b, D_MODEL * D_MODEL);
  cvt(wk, wk_b, D_MODEL * D_MODEL);
  cvt(wv, wv_b, D_MODEL * D_MODEL);
  cvt(wo, wo_b, D_MODEL * D_MODEL);
  cvt(w1, w1_b, D_FF * D_MODEL);
  cvt(w2, w2_b, D_MODEL * D_FF);

  const dim3 blk(256);
  const dim3 gD(M_ROWS / 128, D_MODEL / 128);   // (32, 8)
  const dim3 gF(M_ROWS / 128, D_FF / 128);      // (32, 32)
  const size_t lds = 65536;                     // 4 x 16KB tile buffers

  rmsnorm_bf16_kernel<<<dim3(M_ROWS), blk, 0, stream>>>(x, ln1w, Abuf);
  gemm_bf16_kernel<0><<<gD, blk, lds, stream>>>(Abuf, wq_b, M_ROWS, D_MODEL, D_MODEL, Qbuf, nullptr, nullptr);
  gemm_bf16_kernel<0><<<gD, blk, lds, stream>>>(Abuf, wk_b, M_ROWS, D_MODEL, D_MODEL, Kbuf, nullptr, nullptr);
  gemm_bf16_kernel<0><<<gD, blk, lds, stream>>>(Abuf, wv_b, M_ROWS, D_MODEL, D_MODEL, Vbuf, nullptr, nullptr);
  attn_kernel<<<dim3(SEQ / 128, BATCH * N_HEADS), blk, 0, stream>>>(Qbuf, Kbuf, Vbuf, Abuf);
  gemm_bf16_kernel<2><<<gD, blk, lds, stream>>>(Abuf, wo_b, M_ROWS, D_MODEL, D_MODEL, nullptr, Ybuf, x);
  rmsnorm_bf16_kernel<<<dim3(M_ROWS), blk, 0, stream>>>(Ybuf, ln2w, Abuf);
  gemm_bf16_kernel<1><<<gF, blk, lds, stream>>>(Abuf, w1_b, M_ROWS, D_FF, D_MODEL, Hbuf, nullptr, nullptr);
  gemm_bf16_kernel<2><<<gD, blk, lds, stream>>>(Hbuf, w2_b, M_ROWS, D_MODEL, D_FF, nullptr, out, Ybuf);
}
